// ReID_86311662780811
// MI455X (gfx1250) — matrix operand order repaired and checked by reading, not compiled, not run on hardware
//
#include <hip/hip_runtime.h>
#include <hip/hip_bf16.h>

typedef __attribute__((ext_vector_type(16))) _Float16 v16h;
typedef __attribute__((ext_vector_type(8)))  float    v8f;

#define C_IN   256
#define C_OUT  128
#define H_     270
#define W_     480
#define N_     512
#define KTOT   2304            // C_IN * 9
#define KT     72              // KTOT / 32 (k-tiles per WMMA chain)
#define RT     32              // N_ / 16 row tiles
#define CT     8               // C_OUT / 16 col tiles

// workspace layout (bytes)
#define APACK_HALFS (RT * KT * 32 * 16)                 // 1,179,648 halves
#define BPACK_HALFS (CT * KT * 32 * 16)                 //   294,912 halves
#define APACK_OFF   0
#define BPACK_OFF   (APACK_HALFS * 2)                   // 2,359,296 B (32B aligned)
#define VECS_OFF    (BPACK_OFF + BPACK_HALFS * 2)       // 2,949,120 B

// ---------------------------------------------------------------------------
// Kernel 1: gather zero-padded 3x3xC_IN patches at the 512 centers and write
// them directly in the CDNA5 16-bit A-matrix (16x32) fragment layout:
//   lane = (n%16) + 16*bit3(k%32),  h = (k%8) + 8*bit4(k%32)
// ---------------------------------------------------------------------------
__global__ void pack_a(const float* __restrict__ bm,
                       const int* __restrict__ centers,
                       _Float16* __restrict__ Apack) {
  int tid = blockIdx.x * blockDim.x + threadIdx.x;
  if (tid >= N_ * KTOT) return;
  int n = tid / KTOT;
  int k = tid - n * KTOT;
  int ci = k / 9;
  int r  = k - ci * 9;
  int ky = r / 3;
  int kx = r - ky * 3;
  int cx = centers[2 * n + 0];
  int cy = centers[2 * n + 1];
  int x = cx + kx - 1;
  int y = cy + ky - 1;
  float v = 0.0f;
  if (x >= 0 && x < W_ && y >= 0 && y < H_)
    v = bm[ci * (H_ * W_) + y * W_ + x];
  int rt = n >> 4, lr = n & 15;
  int kt = k >> 5, kr = k & 31;
  int lane = lr + (((kr >> 3) & 1) << 4);
  int h    = (kr & 7) + (((kr >> 4) & 1) << 3);
  Apack[(((rt * KT + kt) * 32 + lane) << 4) + h] = (_Float16)v;
}

// ---------------------------------------------------------------------------
// Kernel 2: reshape conv_w [C_OUT][C_IN*9] into B-matrix (32x16) fragments:
//   lane = (co%16) + 16*bit3(k%32),  h = (k%8) + 8*bit4(k%32)
// ---------------------------------------------------------------------------
__global__ void pack_b(const float* __restrict__ w,
                       _Float16* __restrict__ Bpack) {
  int tid = blockIdx.x * blockDim.x + threadIdx.x;
  if (tid >= KTOT * C_OUT) return;
  int co = tid / KTOT;
  int k  = tid - co * KTOT;
  int ct = co >> 4, nn = co & 15;
  int kt = k >> 5,  kr = k & 31;
  int lane = nn + (((kr >> 3) & 1) << 4);
  int h    = (kr & 7) + (((kr >> 4) & 1) << 3);
  Bpack[(((ct * KT + kt) * 32 + lane) << 4) + h] = (_Float16)w[tid];
}

// ---------------------------------------------------------------------------
// Kernel 3: C[512x128] = A x B via v_wmma_f32_16x16x32_f16, + bias + ReLU.
// 8 waves/block, one 16x16 output tile per wave, 72-step K chain with two
// accumulators for ILP. EXEC is all-ones (no divergence around WMMA).
// ---------------------------------------------------------------------------
__global__ void __launch_bounds__(256)
gemm_wmma(const _Float16* __restrict__ Apack,
          const _Float16* __restrict__ Bpack,
          const float* __restrict__ conv_b,
          float* __restrict__ vecs) {
  int wave = threadIdx.x >> 5;
  int lane = threadIdx.x & 31;
  int tile = blockIdx.x * 8 + wave;
  int rt = tile >> 3;          // / CT
  int ct = tile & 7;

  const v16h* Ap = (const v16h*)Apack + (size_t)rt * KT * 32 + lane;
  const v16h* Bp = (const v16h*)Bpack + (size_t)ct * KT * 32 + lane;

  v8f acc0 = {};
  v8f acc1 = {};
#pragma unroll 4
  for (int kt = 0; kt < KT; kt += 2) {
    v16h a0 = Ap[(kt + 0) * 32];
    v16h b0 = Bp[(kt + 0) * 32];
    v16h a1 = Ap[(kt + 1) * 32];
    v16h b1 = Bp[(kt + 1) * 32];
    acc0 = __builtin_amdgcn_wmma_f32_16x16x32_f16(false, a0, false, b0,
                                                  (short)0, acc0, false, false);
    acc1 = __builtin_amdgcn_wmma_f32_16x16x32_f16(false, a1, false, b1,
                                                  (short)0, acc1, false, false);
  }

  // 32-bit C/D layout: M = r + 8*(lane>=16), N = lane%16
  int hi = lane >> 4;
  int nB = lane & 15;
  int co = ct * 16 + nB;
  float bias = conv_b[co];
  float* out = vecs + (size_t)(rt * 16 + hi * 8) * C_OUT + co;
#pragma unroll
  for (int r2 = 0; r2 < 8; ++r2) {
    float v = acc0[r2] + acc1[r2] + bias;
    out[(size_t)r2 * C_OUT] = v > 0.0f ? v : 0.0f;
  }
}

// ---------------------------------------------------------------------------
// Kernel 4: FC(128->5) + stable softmax + NLL reduction. One 512-thread block.
// ---------------------------------------------------------------------------
__global__ void __launch_bounds__(512)
head(const float* __restrict__ vecs,
     const float* __restrict__ L,
     const float* __restrict__ fc_w,
     const float* __restrict__ fc_b,
     float* __restrict__ out) {
  __shared__ float red[512];
  int n = threadIdx.x;

  float lg[5];
#pragma unroll
  for (int j = 0; j < 5; ++j) lg[j] = fc_b[j];

  const float* v = vecs + (size_t)n * C_OUT;
  for (int c = 0; c < C_OUT; ++c) {
    float x = v[c];
#pragma unroll
    for (int j = 0; j < 5; ++j) lg[j] = fmaf(x, fc_w[j * C_OUT + c], lg[j]);
  }

  float m = lg[0];
#pragma unroll
  for (int j = 1; j < 5; ++j) m = fmaxf(m, lg[j]);
  float s = 0.0f;
#pragma unroll
  for (int j = 0; j < 5; ++j) s += expf(lg[j] - m);
  float inv  = 1.0f / s;
  float logs = logf(s);

  float loss = 0.0f;
#pragma unroll
  for (int j = 0; j < 5; ++j) {
    float p = expf(lg[j] - m) * inv;
    out[n * 5 + j] = p;
    loss -= L[n * 5 + j] * (lg[j] - m - logs);   // exact log-softmax
  }

  red[n] = loss;
  __syncthreads();
  for (int st = 256; st > 0; st >>= 1) {
    if (n < st) red[n] += red[n + st];
    __syncthreads();
  }
  if (n == 0) out[N_ * 5] = red[0];
}

// ---------------------------------------------------------------------------
extern "C" void kernel_launch(void* const* d_in, const int* in_sizes, int n_in,
                              void* d_out, int out_size, void* d_ws, size_t ws_size,
                              hipStream_t stream) {
  (void)in_sizes; (void)n_in; (void)out_size; (void)ws_size;
  const float* bm      = (const float*)d_in[0];
  const int*   centers = (const int*)d_in[1];
  const float* L       = (const float*)d_in[2];
  const float* conv_w  = (const float*)d_in[3];
  const float* conv_b  = (const float*)d_in[4];
  const float* fc_w    = (const float*)d_in[5];
  const float* fc_b    = (const float*)d_in[6];
  float* out = (float*)d_out;

  char* ws = (char*)d_ws;
  _Float16* Apack = (_Float16*)(ws + APACK_OFF);
  _Float16* Bpack = (_Float16*)(ws + BPACK_OFF);
  float*    vecs  = (float*)(ws + VECS_OFF);

  pack_a<<<(N_ * KTOT + 255) / 256, 256, 0, stream>>>(bm, centers, Apack);
  pack_b<<<(KTOT * C_OUT + 255) / 256, 256, 0, stream>>>(conv_w, Bpack);
  gemm_wmma<<<(RT * CT) / 8, 256, 0, stream>>>(Apack, Bpack, conv_b, vecs);
  head<<<1, 512, 0, stream>>>(vecs, L, fc_w, fc_b, out);
}
